// SIMMIMVisionTransformer_73693048864982
// MI455X (gfx1250) — compile-verified, matrix operand order checked by reading, not run on hardware
//
#include <hip/hip_runtime.h>
#include <hip/hip_bf16.h>
#include <stdint.h>

#define DIMC 768
#define HEADS 12
#define HD 64
#define NTOK 1025
#define NPAD 1056
#define GRIDN 32
#define DEPTH 12
#define NUM_REL 3972

typedef __attribute__((ext_vector_type(16))) _Float16 v16h;
typedef __attribute__((ext_vector_type(8)))  float    v8f;

union FragA { v16h v; uint32_t u[8]; _Float16 h[16]; };

__device__ __forceinline__ int imin(int a, int b) { return a < b ? a : b; }

// K-index mapping for 16-bit A/B fragments (ISA 7.12.2, 16-bit A-matrix 16x32)
__device__ __forceinline__ int pairK(int j, int half) {
  return (j < 4) ? (half * 8 + 2 * j) : (16 + half * 8 + 2 * (j - 4));
}
__device__ __forceinline__ float geluf(float x) {
  return 0.5f * x * (1.0f + erff(x * 0.70710678118654752f));
}
__device__ __forceinline__ float hmax16(float x) {
  x = fmaxf(x, __shfl_xor(x, 8, 32));
  x = fmaxf(x, __shfl_xor(x, 4, 32));
  x = fmaxf(x, __shfl_xor(x, 2, 32));
  x = fmaxf(x, __shfl_xor(x, 1, 32));
  return x;
}
__device__ __forceinline__ float hsum16(float x) {
  x += __shfl_xor(x, 8, 32);
  x += __shfl_xor(x, 4, 32);
  x += __shfl_xor(x, 2, 32);
  x += __shfl_xor(x, 1, 32);
  return x;
}

enum { EPI_PATCH = 0, EPI_QKV, EPI_PROJ, EPI_FC1, EPI_FC2, EPI_DC1, EPI_DCNCHW };

struct EpiParams {
  const float* bias;
  const float* pos;
  float* t;
  _Float16* out16;
  _Float16* qb; _Float16* kb; _Float16* vt;
  const float* bn_mean; const float* bn_var; const float* bn_w; const float* bn_b;
  float* outf;
  float* o3;
  int di, dj, Hin, Win;
};

// ---------------------------------------------------------------------------
// Generic WMMA GEMM: C[M,N] = A[M,K](f16) x W[N,K]^T(f16), f32 accumulate.
// One wave -> 64(M) x 16(N) tile; block = 8 waves over N.
// ---------------------------------------------------------------------------
template <int EPI>
__global__ __launch_bounds__(256) void gemm_wmma(
    const _Float16* __restrict__ A, const _Float16* __restrict__ W,
    int M, int N, int K, EpiParams p)
{
  const int wave = threadIdx.x >> 5;
  const int lane = threadIdx.x & 31;
  const int half = lane >> 4;
  const int r    = lane & 15;
  const int m0   = blockIdx.x * 64;
  const int n0   = (blockIdx.y * 8 + wave) * 16;
  if (n0 >= N) return;

  v8f zero8 = {0.f, 0.f, 0.f, 0.f, 0.f, 0.f, 0.f, 0.f};
  v8f acc[4];
#pragma unroll
  for (int i = 0; i < 4; ++i) acc[i] = zero8;

  size_t rowA[4];
#pragma unroll
  for (int i = 0; i < 4; ++i)
    rowA[i] = (size_t)imin(m0 + i * 16 + r, M - 1) * (size_t)K;
  const size_t rowW = (size_t)(n0 + r) * (size_t)K;

  for (int k0 = 0; k0 < K; k0 += 32) {
    FragA b;
#pragma unroll
    for (int j = 0; j < 8; ++j)
      b.u[j] = *(const uint32_t*)(W + rowW + k0 + pairK(j, half));
#pragma unroll
    for (int i = 0; i < 4; ++i) {
      FragA a;
#pragma unroll
      for (int j = 0; j < 8; ++j)
        a.u[j] = *(const uint32_t*)(A + rowA[i] + k0 + pairK(j, half));
      acc[i] = __builtin_amdgcn_wmma_f32_16x16x32_f16(
          false, a.v, false, b.v, (short)0, acc[i], false, false);
    }
  }

#pragma unroll
  for (int it = 0; it < 4; ++it) {
#pragma unroll
    for (int v = 0; v < 8; ++v) {
      int row = m0 + it * 16 + half * 8 + v;
      if (row >= M) continue;
      int col = n0 + r;
      float val = acc[it][v];
      if (EPI == EPI_PATCH) {
        int b = row >> 10, hw = row & 1023;
        val += p.bias[col] + p.pos[(size_t)(hw + 1) * DIMC + col];
        p.t[((size_t)(b * NTOK + 1 + hw)) * DIMC + col] = val;
      } else if (EPI == EPI_QKV) {
        val += p.bias[col];
        int b = row / NTOK, tok = row % NTOK;
        int sec = col / DIMC, cc = col % DIMC, head = cc >> 6, d = cc & 63;
        int bh = b * HEADS + head;
        if (sec == 0)
          p.qb[((size_t)bh * NPAD + tok) * HD + d] = (_Float16)val;
        else if (sec == 1)
          p.kb[((size_t)bh * NPAD + tok) * HD + d] = (_Float16)val;
        else
          p.vt[((size_t)bh * HD + d) * NPAD + tok] = (_Float16)val;
      } else if (EPI == EPI_PROJ) {
        size_t o = (size_t)row * DIMC + col;
        p.t[o] = p.t[o] + val + p.bias[col];
      } else if (EPI == EPI_FC1) {
        val = geluf(val + p.bias[col]);
        p.out16[(size_t)row * (4 * DIMC) + col] = (_Float16)val;
      } else if (EPI == EPI_FC2) {
        size_t o = (size_t)row * DIMC + col;
        val += p.bias[col] + p.t[o];
        p.t[o] = val;
        int b = row / NTOK, tok = row % NTOK;
        if (p.out16 && tok >= 1)
          p.out16[((size_t)(b * 1024 + tok - 1)) * DIMC + col] = (_Float16)val;
        if (p.o3 && tok >= 1) {
          int hw = tok - 1;
          p.o3[(((size_t)(b * DIMC + col)) * GRIDN + (hw >> 5)) * GRIDN + (hw & 31)] = val;
        }
      } else if (EPI == EPI_DC1) {
        val += p.bias[col];
        val = (val - p.bn_mean[col]) * rsqrtf(p.bn_var[col] + 1e-5f) * p.bn_w[col] + p.bn_b[col];
        val = geluf(val);
        int b = row >> 10, hw = row & 1023, h = hw >> 5, w = hw & 31;
        size_t orow = (size_t)b * 4096 + (size_t)(2 * h + p.di) * 64 + (2 * w + p.dj);
        p.out16[orow * DIMC + col] = (_Float16)val;
      } else if (EPI == EPI_DCNCHW) {
        val += p.bias[col];
        int hwN = p.Hin * p.Win;
        int b = row / hwN, hw = row % hwN, h = hw / p.Win, w = hw % p.Win;
        int H = 2 * h + p.di, Wc = 2 * w + p.dj;
        int Ho = 2 * p.Hin, Wo = 2 * p.Win;
        p.outf[(((size_t)(b * DIMC + col)) * Ho + H) * Wo + Wc] = val;
      }
    }
  }
}

// ---------------------------------------------------------------------------
// Fused flash attention: per wave = 16 query rows of one (b, head).
// S = (Q*scale)·K^T + relpos  -> online softmax -> O += P·V, all via WMMA.
// ---------------------------------------------------------------------------
__global__ __launch_bounds__(256) void attn_kernel(
    const _Float16* __restrict__ qb, const _Float16* __restrict__ kb,
    const _Float16* __restrict__ vt, const float* __restrict__ relt,
    const int* __restrict__ relidx, _Float16* __restrict__ out)
{
  __shared__ _Float16 Pl[8][16 * 32];
  const int wave = threadIdx.x >> 5;
  const int lane = threadIdx.x & 31;
  const int half = lane >> 4;
  const int r    = lane & 15;
  const int qt   = blockIdx.x * 8 + wave;
  const int head = blockIdx.y;
  const int b    = blockIdx.z;
  const int m0   = qt * 16;
  if (m0 >= NTOK) return;
  const int bh = b * HEADS + head;
  const _Float16* Q  = qb + (size_t)bh * NPAD * HD;
  const _Float16* Kx = kb + (size_t)bh * NPAD * HD;
  const _Float16* V  = vt + (size_t)bh * HD * NPAD;

  FragA aq0, aq1;
  {
    size_t qrow = (size_t)imin(m0 + r, NTOK - 1) * HD;
#pragma unroll
    for (int j = 0; j < 8; ++j) {
      aq0.u[j] = *(const uint32_t*)(Q + qrow + pairK(j, half));
      aq1.u[j] = *(const uint32_t*)(Q + qrow + 32 + pairK(j, half));
    }
  }

  v8f zero8 = {0.f, 0.f, 0.f, 0.f, 0.f, 0.f, 0.f, 0.f};
  float rowmax[8], rowsum[8];
  v8f acc[4];
#pragma unroll
  for (int v = 0; v < 8; ++v) { rowmax[v] = -1e30f; rowsum[v] = 0.f; }
#pragma unroll
  for (int nt = 0; nt < 4; ++nt) acc[nt] = zero8;

  _Float16* P = Pl[wave];

  for (int j0 = 0; j0 < NTOK; j0 += 32) {
    v8f s0 = zero8, s1 = zero8;
    {
      FragA bk;
      size_t kr0 = (size_t)imin(j0 + r, NPAD - 1) * HD;
#pragma unroll
      for (int j = 0; j < 8; ++j) bk.u[j] = *(const uint32_t*)(Kx + kr0 + pairK(j, half));
      s0 = __builtin_amdgcn_wmma_f32_16x16x32_f16(false, aq0.v, false, bk.v, (short)0, s0, false, false);
#pragma unroll
      for (int j = 0; j < 8; ++j) bk.u[j] = *(const uint32_t*)(Kx + kr0 + 32 + pairK(j, half));
      s0 = __builtin_amdgcn_wmma_f32_16x16x32_f16(false, aq1.v, false, bk.v, (short)0, s0, false, false);
      size_t kr1 = (size_t)imin(j0 + 16 + r, NPAD - 1) * HD;
#pragma unroll
      for (int j = 0; j < 8; ++j) bk.u[j] = *(const uint32_t*)(Kx + kr1 + pairK(j, half));
      s1 = __builtin_amdgcn_wmma_f32_16x16x32_f16(false, aq0.v, false, bk.v, (short)0, s1, false, false);
#pragma unroll
      for (int j = 0; j < 8; ++j) bk.u[j] = *(const uint32_t*)(Kx + kr1 + 32 + pairK(j, half));
      s1 = __builtin_amdgcn_wmma_f32_16x16x32_f16(false, aq1.v, false, bk.v, (short)0, s1, false, false);
    }
    const int kk0 = j0 + r, kk1 = j0 + 16 + r;
#pragma unroll
    for (int v = 0; v < 8; ++v) {
      int q = m0 + half * 8 + v;
      float x0 = s0[v] * 0.125f, x1 = s1[v] * 0.125f;
      if (q < NTOK) {
        if (kk0 < NTOK)
          x0 += relt[(size_t)relidx[(size_t)q * NTOK + kk0] * HEADS + head];
        else
          x0 = -1e30f;
        if (kk1 < NTOK)
          x1 += relt[(size_t)relidx[(size_t)q * NTOK + kk1] * HEADS + head];
        else
          x1 = -1e30f;
      }
      float tm = hmax16(fmaxf(x0, x1));
      float nm = fmaxf(rowmax[v], tm);
      float alpha = __expf(rowmax[v] - nm);
      rowmax[v] = nm;
      float e0 = __expf(x0 - nm), e1 = __expf(x1 - nm);
      rowsum[v] = rowsum[v] * alpha + hsum16(e0 + e1);
#pragma unroll
      for (int nt = 0; nt < 4; ++nt) acc[nt][v] *= alpha;
      P[(half * 8 + v) * 32 + r]      = (_Float16)e0;
      P[(half * 8 + v) * 32 + 16 + r] = (_Float16)e1;
    }
    __asm__ volatile("" ::: "memory");  // keep DS write->read order within wave
    FragA ap;
#pragma unroll
    for (int j = 0; j < 8; ++j)
      ap.u[j] = *(const uint32_t*)(P + r * 32 + pairK(j, half));
#pragma unroll
    for (int nt = 0; nt < 4; ++nt) {
      FragA bv;
#pragma unroll
      for (int j = 0; j < 8; ++j)
        bv.u[j] = *(const uint32_t*)(V + (size_t)(nt * 16 + r) * NPAD + j0 + pairK(j, half));
      acc[nt] = __builtin_amdgcn_wmma_f32_16x16x32_f16(
          false, ap.v, false, bv.v, (short)0, acc[nt], false, false);
    }
    __asm__ volatile("" ::: "memory");
  }

#pragma unroll
  for (int v = 0; v < 8; ++v) {
    int q = m0 + half * 8 + v;
    if (q >= NTOK) continue;
    float inv = 1.0f / rowsum[v];
#pragma unroll
    for (int nt = 0; nt < 4; ++nt)
      out[((size_t)(b * NTOK + q)) * DIMC + head * HD + nt * 16 + r] =
          (_Float16)(acc[nt][v] * inv);
  }
}

// ---------------------------------------------------------------------------
// Small utility kernels
// ---------------------------------------------------------------------------
__global__ void cvt_f16(const float* __restrict__ in, _Float16* __restrict__ outp, size_t n) {
  size_t i = (size_t)blockIdx.x * 256 + threadIdx.x;
  if (i < n) outp[i] = (_Float16)in[i];
}
__global__ void zero16(_Float16* p, size_t n) {
  size_t i = (size_t)blockIdx.x * 256 + threadIdx.x;
  if (i < n) p[i] = (_Float16)0.0f;
}
// pack deconv weight [Cin,Cout,2,2] (f32) -> [ij][Cout][Cin] (f16)
__global__ void pack_deconv(const float* __restrict__ w, _Float16* __restrict__ outp) {
  int idx = blockIdx.x * 256 + threadIdx.x;
  if (idx >= 4 * DIMC * DIMC) return;
  int ij = idx / (DIMC * DIMC);
  int rem = idx % (DIMC * DIMC);
  int d = rem / DIMC, c = rem % DIMC;
  int i = ij >> 1, j = ij & 1;
  outp[idx] = (_Float16)w[(((size_t)c * DIMC + d) * 2 + i) * 2 + j];
}
// patchify x[2,3,512,512] -> A0[2048, 768] f16 (col = c*256 + i*16 + j)
__global__ void patchify(const float* __restrict__ x, _Float16* __restrict__ A0) {
  int idx = blockIdx.x * 256 + threadIdx.x;
  if (idx >= 2048 * DIMC) return;
  int col = idx % DIMC, row = idx / DIMC;
  int b = row >> 10, hw = row & 1023, gh = hw >> 5, gw = hw & 31;
  int c = col >> 8, rem = col & 255, i = rem >> 4, j = rem & 15;
  A0[idx] = (_Float16)x[(((size_t)(b * 3 + c) * 512) + gh * 16 + i) * 512 + gw * 16 + j];
}
__global__ void cls_kernel(const float* __restrict__ cls, const float* __restrict__ pos,
                           float* __restrict__ t) {
  int i = blockIdx.x * 256 + threadIdx.x;
  if (i >= 2 * DIMC) return;
  int b = i / DIMC, c = i % DIMC;
  t[(size_t)(b * NTOK) * DIMC + c] = cls[c] + pos[c];
}
__global__ void build_qkv_bias(const float* __restrict__ qbias, const float* __restrict__ vbias,
                               float* __restrict__ outp, int layer) {
  int c = blockIdx.x * 256 + threadIdx.x;
  if (c >= 3 * DIMC) return;
  float v = 0.f;
  if (c < DIMC) v = qbias[layer * DIMC + c];
  else if (c >= 2 * DIMC) v = vbias[layer * DIMC + c - 2 * DIMC];
  outp[c] = v;
}
__global__ __launch_bounds__(256) void ln_kernel(
    const float* __restrict__ t, const float* __restrict__ w, const float* __restrict__ b,
    _Float16* __restrict__ outp) {
  int row = blockIdx.x;
  int tid = threadIdx.x;
  const float* xr = t + (size_t)row * DIMC;
  float l0 = xr[tid], l1 = xr[tid + 256], l2 = xr[tid + 512];
  __shared__ float red[8];
  float s = l0 + l1 + l2;
  for (int m = 16; m >= 1; m >>= 1) s += __shfl_xor(s, m, 32);
  if ((tid & 31) == 0) red[tid >> 5] = s;
  __syncthreads();
  float tot = 0.f;
  for (int i = 0; i < 8; ++i) tot += red[i];
  float mean = tot * (1.0f / 768.0f);
  float d0 = l0 - mean, d1 = l1 - mean, d2 = l2 - mean;
  float s2 = d0 * d0 + d1 * d1 + d2 * d2;
  __syncthreads();
  for (int m = 16; m >= 1; m >>= 1) s2 += __shfl_xor(s2, m, 32);
  if ((tid & 31) == 0) red[tid >> 5] = s2;
  __syncthreads();
  float tv = 0.f;
  for (int i = 0; i < 8; ++i) tv += red[i];
  float rs = rsqrtf(tv * (1.0f / 768.0f) + 1e-6f);
  _Float16* o = outp + (size_t)row * DIMC;
  o[tid]       = (_Float16)(d0 * rs * w[tid] + b[tid]);
  o[tid + 256] = (_Float16)(d1 * rs * w[tid + 256] + b[tid + 256]);
  o[tid + 512] = (_Float16)(d2 * rs * w[tid + 512] + b[tid + 512]);
}
__global__ void maxpool_kernel(const _Float16* __restrict__ feat, float* __restrict__ outp) {
  int idx = blockIdx.x * 256 + threadIdx.x;
  if (idx >= 2 * DIMC * 16 * 16) return;
  int w = idx & 15;
  int h = (idx >> 4) & 15;
  int rem = idx >> 8;
  int d = rem % DIMC;
  int b = rem / DIMC;
  float m = -1e30f;
  for (int di = 0; di < 2; ++di)
    for (int dj = 0; dj < 2; ++dj) {
      int tok = (2 * h + di) * GRIDN + (2 * w + dj);
      m = fmaxf(m, (float)feat[((size_t)(b * 1024 + tok)) * DIMC + d]);
    }
  outp[(((size_t)(b * DIMC + d)) * 16 + h) * 16 + w] = m;
}

// ---------------------------------------------------------------------------
extern "C" void kernel_launch(void* const* d_in, const int* in_sizes, int n_in,
                              void* d_out, int out_size, void* d_ws, size_t ws_size,
                              hipStream_t stream) {
  const float* x        = (const float*)d_in[0];
  const float* patch_w  = (const float*)d_in[1];
  const float* patch_b  = (const float*)d_in[2];
  const float* cls_tok  = (const float*)d_in[3];
  const float* pos      = (const float*)d_in[4];
  const float* ln1_w    = (const float*)d_in[5];
  const float* ln1_b    = (const float*)d_in[6];
  const float* qkv_w    = (const float*)d_in[7];
  const float* q_bias   = (const float*)d_in[8];
  const float* v_bias   = (const float*)d_in[9];
  const float* rel_tab  = (const float*)d_in[10];
  const float* proj_w   = (const float*)d_in[11];
  const float* proj_b   = (const float*)d_in[12];
  const float* ln2_w    = (const float*)d_in[13];
  const float* ln2_b    = (const float*)d_in[14];
  const float* fc1_w    = (const float*)d_in[15];
  const float* fc1_b    = (const float*)d_in[16];
  const float* fc2_w    = (const float*)d_in[17];
  const float* fc2_b    = (const float*)d_in[18];
  const float* d1_w     = (const float*)d_in[19];
  const float* d1_b     = (const float*)d_in[20];
  const float* bn_w     = (const float*)d_in[21];
  const float* bn_b     = (const float*)d_in[22];
  const float* bn_mean  = (const float*)d_in[23];
  const float* bn_var   = (const float*)d_in[24];
  const float* d2_w     = (const float*)d_in[25];
  const float* d2_b     = (const float*)d_in[26];
  const float* f2_w     = (const float*)d_in[27];
  const float* f2_b     = (const float*)d_in[28];
  const int*   rel_idx  = (const int*)d_in[29];
  float* out = (float*)d_out;

  char* base = (char*)d_ws;
  size_t off = 0;
  auto alloc = [&](size_t bytes) -> void* {
    void* p = base + off;
    off = (off + bytes + 255) & ~(size_t)255;
    return p;
  };

  _Float16* w_qkv16  = (_Float16*)alloc((size_t)DEPTH * 3 * DIMC * DIMC * 2);
  _Float16* w_proj16 = (_Float16*)alloc((size_t)DEPTH * DIMC * DIMC * 2);
  _Float16* w_fc116  = (_Float16*)alloc((size_t)DEPTH * 4 * DIMC * DIMC * 2);
  _Float16* w_fc216  = (_Float16*)alloc((size_t)DEPTH * 4 * DIMC * DIMC * 2);
  _Float16* w_patch  = (_Float16*)alloc((size_t)DIMC * DIMC * 2);
  _Float16* w_d1p    = (_Float16*)alloc((size_t)4 * DIMC * DIMC * 2);
  _Float16* w_d2p    = (_Float16*)alloc((size_t)4 * DIMC * DIMC * 2);
  _Float16* w_f2p    = (_Float16*)alloc((size_t)4 * DIMC * DIMC * 2);
  _Float16* A0       = (_Float16*)alloc((size_t)2048 * DIMC * 2);
  float*    t        = (float*)alloc((size_t)2 * NTOK * DIMC * 4);
  _Float16* h        = (_Float16*)alloc((size_t)2 * NTOK * DIMC * 2);
  float*    qkvbias  = (float*)alloc((size_t)3 * DIMC * 4);
  _Float16* qbuf     = (_Float16*)alloc((size_t)2 * HEADS * NPAD * HD * 2);
  _Float16* kbuf     = (_Float16*)alloc((size_t)2 * HEADS * NPAD * HD * 2);
  _Float16* vtbuf    = (_Float16*)alloc((size_t)2 * HEADS * HD * NPAD * 2);
  _Float16* attn16   = (_Float16*)alloc((size_t)2 * NTOK * DIMC * 2);
  _Float16* mbuf     = (_Float16*)alloc((size_t)2 * NTOK * 4 * DIMC * 2);
  _Float16* feat0    = (_Float16*)alloc((size_t)2048 * DIMC * 2);
  _Float16* feat1    = (_Float16*)alloc((size_t)2048 * DIMC * 2);
  _Float16* feat3    = (_Float16*)alloc((size_t)2048 * DIMC * 2);
  _Float16* ybuf     = (_Float16*)alloc((size_t)8192 * DIMC * 2);
  if (off > ws_size) return;  // workspace too small: deterministic no-op

  auto cvt = [&](const float* src, _Float16* dst, size_t n) {
    cvt_f16<<<(unsigned)((n + 255) / 256), 256, 0, stream>>>(src, dst, n);
  };
  cvt(qkv_w,  w_qkv16,  (size_t)DEPTH * 3 * DIMC * DIMC);
  cvt(proj_w, w_proj16, (size_t)DEPTH * DIMC * DIMC);
  cvt(fc1_w,  w_fc116,  (size_t)DEPTH * 4 * DIMC * DIMC);
  cvt(fc2_w,  w_fc216,  (size_t)DEPTH * 4 * DIMC * DIMC);
  cvt(patch_w, w_patch, (size_t)DIMC * DIMC);
  pack_deconv<<<(4 * DIMC * DIMC + 255) / 256, 256, 0, stream>>>(d1_w, w_d1p);
  pack_deconv<<<(4 * DIMC * DIMC + 255) / 256, 256, 0, stream>>>(d2_w, w_d2p);
  pack_deconv<<<(4 * DIMC * DIMC + 255) / 256, 256, 0, stream>>>(f2_w, w_f2p);
  patchify<<<(2048 * DIMC + 255) / 256, 256, 0, stream>>>(x, A0);

  size_t nz = (size_t)2 * HEADS * NPAD * HD;
  zero16<<<(unsigned)((nz + 255) / 256), 256, 0, stream>>>(qbuf, nz);
  zero16<<<(unsigned)((nz + 255) / 256), 256, 0, stream>>>(kbuf, nz);
  zero16<<<(unsigned)((nz + 255) / 256), 256, 0, stream>>>(vtbuf, nz);

  // patch embed GEMM + cls row
  {
    EpiParams e{}; e.bias = patch_b; e.pos = pos; e.t = t;
    gemm_wmma<EPI_PATCH><<<dim3(2048 / 64, DIMC / 128), 256, 0, stream>>>(
        A0, w_patch, 2048, DIMC, DIMC, e);
    cls_kernel<<<(2 * DIMC + 255) / 256, 256, 0, stream>>>(cls_tok, pos, t);
  }

  const int Mrows = 2 * NTOK;  // 2050
  const int MB = (Mrows + 63) / 64;  // 33

  for (int L = 0; L < DEPTH; ++L) {
    build_qkv_bias<<<(3 * DIMC + 255) / 256, 256, 0, stream>>>(q_bias, v_bias, qkvbias, L);
    ln_kernel<<<Mrows, 256, 0, stream>>>(t, ln1_w + (size_t)L * DIMC, ln1_b + (size_t)L * DIMC, h);
    {
      EpiParams e{}; e.bias = qkvbias; e.qb = qbuf; e.kb = kbuf; e.vt = vtbuf;
      gemm_wmma<EPI_QKV><<<dim3(MB, (3 * DIMC) / 128), 256, 0, stream>>>(
          h, w_qkv16 + (size_t)L * 3 * DIMC * DIMC, Mrows, 3 * DIMC, DIMC, e);
    }
    attn_kernel<<<dim3(9, HEADS, 2), 256, 0, stream>>>(
        qbuf, kbuf, vtbuf, rel_tab + (size_t)L * NUM_REL * HEADS, rel_idx, attn16);
    {
      EpiParams e{}; e.bias = proj_b + (size_t)L * DIMC; e.t = t;
      gemm_wmma<EPI_PROJ><<<dim3(MB, DIMC / 128), 256, 0, stream>>>(
          attn16, w_proj16 + (size_t)L * DIMC * DIMC, Mrows, DIMC, DIMC, e);
    }
    ln_kernel<<<Mrows, 256, 0, stream>>>(t, ln2_w + (size_t)L * DIMC, ln2_b + (size_t)L * DIMC, h);
    {
      EpiParams e{}; e.bias = fc1_b + (size_t)L * 4 * DIMC; e.out16 = mbuf;
      gemm_wmma<EPI_FC1><<<dim3(MB, (4 * DIMC) / 128), 256, 0, stream>>>(
          h, w_fc116 + (size_t)L * 4 * DIMC * DIMC, Mrows, 4 * DIMC, DIMC, e);
    }
    {
      EpiParams e{}; e.bias = fc2_b + (size_t)L * DIMC; e.t = t;
      e.out16 = (L == 3) ? feat0 : (L == 5) ? feat1 : (L == 11) ? feat3 : nullptr;
      e.o3 = (L == 7) ? (out + 31457280) : nullptr;
      gemm_wmma<EPI_FC2><<<dim3(MB, DIMC / 128), 256, 0, stream>>>(
          mbuf, w_fc216 + (size_t)L * 4 * DIMC * DIMC, Mrows, DIMC, 4 * DIMC, e);
    }
  }

  // FPN heads
  for (int ij = 0; ij < 4; ++ij) {  // fpn1 stage 1: deconv + BN + GELU -> ybuf
    EpiParams e{}; e.bias = d1_b; e.bn_mean = bn_mean; e.bn_var = bn_var;
    e.bn_w = bn_w; e.bn_b = bn_b; e.out16 = ybuf; e.di = ij >> 1; e.dj = ij & 1;
    gemm_wmma<EPI_DC1><<<dim3(2048 / 64, DIMC / 128), 256, 0, stream>>>(
        feat0, w_d1p + (size_t)ij * DIMC * DIMC, 2048, DIMC, DIMC, e);
  }
  for (int ij = 0; ij < 4; ++ij) {  // fpn1 stage 2 -> o1 [2,768,128,128]
    EpiParams e{}; e.bias = d2_b; e.outf = out; e.di = ij >> 1; e.dj = ij & 1;
    e.Hin = 64; e.Win = 64;
    gemm_wmma<EPI_DCNCHW><<<dim3(8192 / 64, DIMC / 128), 256, 0, stream>>>(
        ybuf, w_d2p + (size_t)ij * DIMC * DIMC, 8192, DIMC, DIMC, e);
  }
  for (int ij = 0; ij < 4; ++ij) {  // fpn2 -> o2 [2,768,64,64]
    EpiParams e{}; e.bias = f2_b; e.outf = out + 25165824; e.di = ij >> 1; e.dj = ij & 1;
    e.Hin = 32; e.Win = 32;
    gemm_wmma<EPI_DCNCHW><<<dim3(2048 / 64, DIMC / 128), 256, 0, stream>>>(
        feat1, w_f2p + (size_t)ij * DIMC * DIMC, 2048, DIMC, DIMC, e);
  }
  // o4: 2x2 maxpool of feat3 -> [2,768,16,16]
  maxpool_kernel<<<(2 * DIMC * 256 + 255) / 256, 256, 0, stream>>>(feat3, out + 33030144);
}
